// SAGE_8237747274005
// MI455X (gfx1250) — compile-verified, matrix operand order checked by reading, not run on hardware
//
#include <hip/hip_runtime.h>

typedef __attribute__((ext_vector_type(16))) __bf16 v16bf;
typedef __attribute__((ext_vector_type(8)))  __bf16 v8bf;
typedef __attribute__((ext_vector_type(8)))  float  v8f;
typedef __attribute__((ext_vector_type(4)))  float  v4f;

#define DH 256
#define N0S 1000000
#define N1S 200000
#define N2S 24576
#define E1S 2000000
#define E2S 245760

// ---------------- weight fp32 -> bf16 conversion ----------------
__global__ void sage_wcvt(const float* __restrict__ w, __bf16* __restrict__ o, int n) {
    int i = blockIdx.x * blockDim.x + threadIdx.x;
    if (i < n) o[i] = (__bf16)w[i];
}

// ---------------- edge gather + scatter-add (mean numerator & counts) ------
// thread = (edge, 16-byte chunk); 64 chunks of float4 cover D=256
__global__ void sage_scatter(const float* __restrict__ xsrc,
                             const int*   __restrict__ src,
                             const int*   __restrict__ dst,
                             float* __restrict__ sums,
                             float* __restrict__ cnt,
                             int nedges) {
    long long gid = (long long)blockIdx.x * blockDim.x + threadIdx.x;
    int e = (int)(gid >> 6);
    int c = (int)(gid & 63);
    if (e >= nedges) return;
    int s = src[e];
    int d = dst[e];
    const float4 v = ((const float4*)(xsrc + (long long)s * DH))[c];
    float* p = sums + (long long)d * DH + c * 4;
    atomicAdd(p + 0, v.x);
    atomicAdd(p + 1, v.y);
    atomicAdd(p + 2, v.z);
    atomicAdd(p + 3, v.w);
    if (c == 0) atomicAdd(&cnt[d], 1.0f);
}

// ---------------- WMMA fragment loaders ----------------
// 16-bit A 16x32 layout (ISA 7.12.2): lane&15 = row; lanes 0-15 hold K=k0+{0..7,16..23},
// lanes 16-31 hold K=k0+{8..15,24..31}. Same striping used for B via W row-major [N,K].
__device__ __forceinline__ v16bf frag_a_f32(const float* __restrict__ row,
                                            int k0, int half, float scale) {
    const float* p = row + k0 + half * 8;
    v4f p0 = *(const v4f*)(p);
    v4f p1 = *(const v4f*)(p + 4);
    v4f p2 = *(const v4f*)(p + 16);
    v4f p3 = *(const v4f*)(p + 20);
    v16bf a;
#pragma unroll
    for (int j = 0; j < 4; ++j) {
        a[j]      = (__bf16)(p0[j] * scale);
        a[4 + j]  = (__bf16)(p1[j] * scale);
        a[8 + j]  = (__bf16)(p2[j] * scale);
        a[12 + j] = (__bf16)(p3[j] * scale);
    }
    return a;
}

__device__ __forceinline__ v16bf frag_b_bf16(const __bf16* __restrict__ row,
                                             int k0, int half) {
    const __bf16* p = row + k0 + half * 8;
    v8bf q0 = *(const v8bf*)(p);
    v8bf q1 = *(const v8bf*)(p + 16);
    v16bf b;
#pragma unroll
    for (int j = 0; j < 8; ++j) { b[j] = q0[j]; b[8 + j] = q1[j]; }
    return b;
}

// ---------------- fused SAGE GEMM: out = relu?( agg@Wl.T + bl + xt@Wr.T ) ----
// block = 256 threads = 8 waves; block owns rows [m0, m0+16) x all 256 cols.
// wave w computes n-tiles n0=w*32 and n0+16. In-place safe (sync before store).
__global__ void sage_gemm(const float* __restrict__ sums,
                          const float* __restrict__ cnt,
                          const float* __restrict__ xt,
                          const __bf16* __restrict__ Wl,
                          const __bf16* __restrict__ Wr,
                          const float* __restrict__ bl,
                          float* __restrict__ out,
                          int relu) {
    const int lane = threadIdx.x & 31;
    const int wave = threadIdx.x >> 5;
    const int half = lane >> 4;
    const int r    = lane & 15;
    const int m0   = blockIdx.x * 16;
    const int row  = m0 + r;
    const int n0   = wave * 32;

    const float inv = 1.0f / fmaxf(cnt[row], 1.0f);
    const float* aggRow = sums + (long long)row * DH;
    const float* xRow   = xt   + (long long)row * DH;
    const __bf16* wl0 = Wl + (long long)(n0 + r) * DH;
    const __bf16* wl1 = Wl + (long long)(n0 + 16 + r) * DH;
    const __bf16* wr0 = Wr + (long long)(n0 + r) * DH;
    const __bf16* wr1 = Wr + (long long)(n0 + 16 + r) * DH;

    v8f c0, c1;
    const float b0 = bl[n0 + r];
    const float b1 = bl[n0 + 16 + r];
#pragma unroll
    for (int v = 0; v < 8; ++v) { c0[v] = b0; c1[v] = b1; }

    for (int k0 = 0; k0 < DH; k0 += 32) {
        v16bf aA = frag_a_f32(aggRow, k0, half, inv);
        v16bf aX = frag_a_f32(xRow,   k0, half, 1.0f);
        v16bf bL0 = frag_b_bf16(wl0, k0, half);
        v16bf bL1 = frag_b_bf16(wl1, k0, half);
        v16bf bR0 = frag_b_bf16(wr0, k0, half);
        v16bf bR1 = frag_b_bf16(wr1, k0, half);
        c0 = __builtin_amdgcn_wmma_f32_16x16x32_bf16(false, aA, false, bL0, (short)0, c0, false, false);
        c0 = __builtin_amdgcn_wmma_f32_16x16x32_bf16(false, aX, false, bR0, (short)0, c0, false, false);
        c1 = __builtin_amdgcn_wmma_f32_16x16x32_bf16(false, aA, false, bL1, (short)0, c1, false, false);
        c1 = __builtin_amdgcn_wmma_f32_16x16x32_bf16(false, aX, false, bR1, (short)0, c1, false, false);
    }

    if (relu) {
#pragma unroll
        for (int v = 0; v < 8; ++v) { c0[v] = fmaxf(c0[v], 0.0f); c1[v] = fmaxf(c1[v], 0.0f); }
    }

    // all waves in block must finish reading A rows before in-place store
    __syncthreads();

    // C layout: col = lane&15 within tile, row = 8*half + vgpr
#pragma unroll
    for (int v = 0; v < 8; ++v) {
        const long long orow = m0 + half * 8 + v;
        out[orow * DH + n0 + r]      = c0[v];
        out[orow * DH + n0 + 16 + r] = c1[v];
    }
}

extern "C" void kernel_launch(void* const* d_in, const int* in_sizes, int n_in,
                              void* d_out, int out_size, void* d_ws, size_t ws_size,
                              hipStream_t stream) {
    const float* x    = (const float*)d_in[0];
    const int*   src1 = (const int*)d_in[1];
    const int*   dst1 = (const int*)d_in[2];
    const int*   src2 = (const int*)d_in[3];
    const int*   dst2 = (const int*)d_in[4];
    const float* Wl1  = (const float*)d_in[5];
    const float* bl1  = (const float*)d_in[6];
    const float* Wr1  = (const float*)d_in[7];
    const float* Wl2  = (const float*)d_in[8];
    const float* bl2  = (const float*)d_in[9];
    const float* Wr2  = (const float*)d_in[10];

    // workspace layout (all offsets 256B aligned)
    char* ws = (char*)d_ws;
    const size_t off_sums1 = 0;
    const size_t sz_sums1  = (size_t)N1S * DH * sizeof(float);        // 204.8 MB
    const size_t off_cnt1  = off_sums1 + sz_sums1;
    const size_t sz_cnt1   = (size_t)N1S * sizeof(float);
    const size_t off_sums2 = off_cnt1 + sz_cnt1;
    const size_t sz_sums2  = (size_t)N2S * DH * sizeof(float);        // 25.2 MB
    const size_t off_cnt2  = off_sums2 + sz_sums2;
    const size_t sz_cnt2   = (size_t)N2S * sizeof(float);
    const size_t zero_bytes = off_cnt2 + sz_cnt2;
    const size_t sz_w      = (size_t)DH * DH * sizeof(__bf16);        // 128 KB each
    const size_t off_wl1   = zero_bytes;
    const size_t off_wr1   = off_wl1 + sz_w;
    const size_t off_wl2   = off_wr1 + sz_w;
    const size_t off_wr2   = off_wl2 + sz_w;

    float*  sums1 = (float*)(ws + off_sums1);
    float*  cnt1  = (float*)(ws + off_cnt1);
    float*  sums2 = (float*)(ws + off_sums2);
    float*  cnt2  = (float*)(ws + off_cnt2);
    __bf16* wl1b  = (__bf16*)(ws + off_wl1);
    __bf16* wr1b  = (__bf16*)(ws + off_wr1);
    __bf16* wl2b  = (__bf16*)(ws + off_wl2);
    __bf16* wr2b  = (__bf16*)(ws + off_wr2);

    // zero accumulators & counts (graph-capturable)
    hipMemsetAsync(ws, 0, zero_bytes, stream);

    // bf16 weight conversion
    const int wn = DH * DH;
    sage_wcvt<<<(wn + 255) / 256, 256, 0, stream>>>(Wl1, wl1b, wn);
    sage_wcvt<<<(wn + 255) / 256, 256, 0, stream>>>(Wr1, wr1b, wn);
    sage_wcvt<<<(wn + 255) / 256, 256, 0, stream>>>(Wl2, wl2b, wn);
    sage_wcvt<<<(wn + 255) / 256, 256, 0, stream>>>(Wr2, wr2b, wn);

    // layer 1: scatter-mean numerator + counts, then fused GEMM (in-place -> h1)
    {
        long long thr = (long long)E1S * 64;
        sage_scatter<<<(unsigned)((thr + 255) / 256), 256, 0, stream>>>(
            x, src1, dst1, sums1, cnt1, E1S);
        sage_gemm<<<N1S / 16, 256, 0, stream>>>(
            sums1, cnt1, x, wl1b, wr1b, bl1, /*out=*/sums1, /*relu=*/1);
    }

    // layer 2: gather from h1 (== sums1), write d_out
    {
        long long thr = (long long)E2S * 64;
        sage_scatter<<<(unsigned)((thr + 255) / 256), 256, 0, stream>>>(
            sums1, src2, dst2, sums2, cnt2, E2S);
        sage_gemm<<<N2S / 16, 256, 0, stream>>>(
            sums2, cnt2, /*xt=*/sums1, wl2b, wr2b, bl2, (float*)d_out, /*relu=*/0);
    }

    (void)in_sizes; (void)n_in; (void)out_size; (void)ws_size;
}
